// SyntheticCNV_32409823215999
// MI455X (gfx1250) — 8-bit matrix operand order repaired and checked by reading, compiled, not run on hardware
//
#include <hip/hip_runtime.h>
#include <stdint.h>

typedef __attribute__((ext_vector_type(8))) int v8i;
typedef __attribute__((ext_vector_type(4))) int v4i;

#define BATCH 32
#define CH    256
#define HW    32
#define PHW   34
#define KTOT  2304
#define KSTEPS 36
#define MTILES 2048
#define NTILES 16

__global__ __launch_bounds__(256) void quant_in_kernel(
    const float* __restrict__ x, int8_t* __restrict__ xq)
{
    const size_t idx = (size_t)blockIdx.x * 256 + threadIdx.x;
    const int c = idx & 255;
    const int w = (idx >> 8) % PHW;
    const int h = (idx >> 8) / PHW % PHW;
    const int b = (int)(idx >> 8) / (PHW * PHW);
    int8_t q = 0;
    if (h >= 1 && h <= HW && w >= 1 && w <= HW) {
        const float v = x[((((size_t)b * CH + c) * HW + (h - 1)) << 5) + (w - 1)];
        float r = rintf(v * 128.0f);
        r = fminf(fmaxf(r, -128.0f), 127.0f);
        q = (int8_t)(int)r;
    }
    xq[idx] = q;
}

__global__ __launch_bounds__(256) void pack_w_kernel(
    const float* __restrict__ w, int8_t* __restrict__ bp)
{
    const int idx = blockIdx.x * 256 + threadIdx.x;
    const int v  = idx & 7;
    const int L  = (idx >> 3) & 31;
    const int ks = (idx >> 8) % KSTEPS;
    const int nt = (idx >> 8) / KSTEPS;
    const int n  = nt * 16 + (L & 15);
    const int hi = L >> 4;
    uint32_t word = 0;
#pragma unroll
    for (int j = 0; j < 4; ++j) {
        const int k  = ks * 64 + (v >> 2) * 32 + hi * 16 + (v & 3) * 4 + j;
        const int kh = k / 768;
        const int kw = (k >> 8) % 3;
        const int c  = k & 255;
        const float wv = w[((((size_t)n * CH + c) * 3 + kh) * 3) + kw];
        float r = fminf(fmaxf(rintf(wv), -1.0f), 1.0f);
        word |= ((uint32_t)(uint8_t)(int8_t)(int)r) << (8 * j);
    }
    ((uint32_t*)bp)[idx] = word;
}

__global__ __launch_bounds__(256) void cnv_wmma_kernel(
    const int8_t* __restrict__ xq, const int8_t* __restrict__ bp,
    const float* __restrict__ gamma, const float* __restrict__ beta,
    const float* __restrict__ rmean, const float* __restrict__ rvar,
    float* __restrict__ out)
{
    const int lane  = threadIdx.x & 31;
    const int wave  = threadIdx.x >> 5;
    const int task  = blockIdx.x * 8 + wave;
    const int mtile = task >> 2;
    const int ng    = task & 3;

    const int hi  = lane >> 4;
    const int mlo = lane & 15;

    const int mrow = mtile * 16 + mlo;
    const int b  = mrow >> 10;
    const int oh = (mrow >> 5) & 31;
    const int ow = mrow & 31;

    v8i acc0 = {}, acc1 = {}, acc2 = {}, acc3 = {};

#pragma unroll 4
    for (int ks = 0; ks < KSTEPS; ++ks) {
        const int tap  = ks >> 2;
        const int cblk = (ks & 3) << 6;
        const int kh   = tap / 3;
        const int kw   = tap - kh * 3;

        const int8_t* aRow = xq +
            ((((size_t)(b * PHW + oh + kh)) * PHW + (ow + kw)) << 8) + cblk;
        union { v8i v; unsigned long long u[4]; } A;
#pragma unroll
        for (int v2 = 0; v2 < 4; ++v2)
            A.u[v2] = *(const unsigned long long*)(aRow + 32 * (v2 >> 1) + 8 * (v2 & 1) + 16 * hi);

        const int8_t* bBase = bp + ((((size_t)(ng * 4) * KSTEPS + ks) << 10) + (lane << 5));
        if (ks + 1 < KSTEPS)
            __builtin_prefetch(bBase + 1024, 0, 0);

        union { v8i v; v4i q[2]; } B;
#pragma unroll
        for (int t = 0; t < 4; ++t) {
            const int8_t* bPtr = bBase + ((size_t)t * KSTEPS << 10);
            B.q[0] = *(const v4i*)(bPtr);
            B.q[1] = *(const v4i*)(bPtr + 16);
            v8i* accp = (t == 0) ? &acc0 : (t == 1) ? &acc1 : (t == 2) ? &acc2 : &acc3;
            *accp = __builtin_amdgcn_wmma_i32_16x16x64_iu8(
                 true, A.v,  true, B.v, *accp,
                 false,  false);
        }
    }

    const float sx = 1.0f / 128.0f;
    v8i accs[4] = { acc0, acc1, acc2, acc3 };
#pragma unroll
    for (int t = 0; t < 4; ++t) {
        const int n = (ng * 4 + t) * 16 + mlo;
        const float inv   = gamma[n] * rsqrtf(rvar[n] + 1e-4f);
        const float scale = inv * sx;
        const float bias  = beta[n] - rmean[n] * inv;
#pragma unroll
        for (int r = 0; r < 8; ++r) {
            const int m   = mtile * 16 + hi * 8 + r;
            const int ob  = m >> 10;
            const int ooh = (m >> 5) & 31;
            const int oow = m & 31;
            const float y = (float)accs[t][r] * scale + bias;
            const float q = fminf(fmaxf(rintf(y * 2.0f), -2.0f), 1.0f) * 0.5f;
            out[((((size_t)ob * CH + n) * HW + ooh) << 5) + oow] = q;
        }
    }
}

extern "C" void kernel_launch(void* const* d_in, const int* in_sizes, int n_in,
                              void* d_out, int out_size, void* d_ws, size_t ws_size,
                              hipStream_t stream)
{
    const float* x     = (const float*)d_in[0];
    const float* w     = (const float*)d_in[1];
    const float* gamma = (const float*)d_in[2];
    const float* beta  = (const float*)d_in[3];
    const float* rmean = (const float*)d_in[4];
    const float* rvar  = (const float*)d_in[5];
    float* out = (float*)d_out;

    const size_t xq_bytes = (size_t)BATCH * PHW * PHW * CH;
    int8_t* xq = (int8_t*)d_ws;
    int8_t* bp = xq + xq_bytes;

    quant_in_kernel<<<(unsigned)(xq_bytes / 256), 256, 0, stream>>>(x, xq);
    pack_w_kernel<<<(NTILES * KSTEPS * 256) / 256, 256, 0, stream>>>(w, bp);
    cnv_wmma_kernel<<<1024, 256, 0, stream>>>(xq, bp, gamma, beta, rmean, rvar, out);
}
